// Encoder_45226005626971
// MI455X (gfx1250) — matrix operand order repaired and checked by reading, not compiled, not run on hardware
//
#include <hip/hip_runtime.h>
#include <hip/hip_bf16.h>
#include <cstdint>

#define N_NODES    100000
#define N_EDGES    1000000
#define DIM        64
#define NUM_GRAPHS 512
#define BN_EPS     1e-5f

typedef __attribute__((ext_vector_type(16))) _Float16     v16h;
typedef __attribute__((ext_vector_type(8)))  float        v8f;
typedef __attribute__((ext_vector_type(4)))  unsigned int u32x4;
typedef __attribute__((ext_vector_type(8)))  int          i32x8;
typedef __attribute__((ext_vector_type(4)))  int          i32x4;

// async-LDS builtin wants vector-typed, address-space-qualified pointers
typedef __attribute__((address_space(1))) i32x4 as1_i32x4;   // global
typedef __attribute__((address_space(3))) i32x4 as3_i32x4;   // LDS (32-bit ptr)

// ---------------- staging-path selection (all __has_builtin guarded) ----
#if defined(__AMDGCN__)
#if __has_builtin(__builtin_amdgcn_tensor_load_to_lds) && \
    __has_builtin(__builtin_amdgcn_s_wait_tensorcnt)
#if __has_include(<hip/amd_detail/amd_gfx1250_TDM.h>)
#define STAGE_TDM6 1         // therock-10.0 6-arg TDM builtin
#else
#define STAGE_TDM5 1         // ROCm 7.2 5-arg TDM builtin
#endif
#elif __has_builtin(__builtin_amdgcn_global_load_async_to_lds_b128) && \
      __has_builtin(__builtin_amdgcn_s_wait_asynccnt)
#define STAGE_ASYNC 1        // async global->LDS copies
#endif
#endif

__device__ __forceinline__ float fast_tanh(float x) {
#if defined(__AMDGCN__) && __has_builtin(__builtin_amdgcn_tanhf)
    return __builtin_amdgcn_tanhf(x);   // v_tanh_f32 (CDNA5 TRANS op)
#else
    return tanhf(x);
#endif
}

// ---------------------------------------------------------------- zero
__global__ void zero_f32(float* __restrict__ p, int n) {
    int i = blockIdx.x * blockDim.x + threadIdx.x;
    if (i < n) p[i] = 0.0f;
}

// ------------------------------------------------- weight fragment prep
// Pre-swizzle the four 64x64 f32 weight matrices into per-lane WMMA
// B-fragment order (f16): frag[m][(ct*2+c)*32 + lane][h] =
//   W[c*32 + (h>>3)*16 + (lane>>4)*8 + (h&7)][ct*16 + (lane&15)]
__global__ void prep_weights(const float* __restrict__ Wa,
                             const float* __restrict__ Wb,
                             const float* __restrict__ Wc,
                             const float* __restrict__ Wd,
                             _Float16* __restrict__ frags) {
    const float* Ws[4] = {Wa, Wb, Wc, Wd};
    for (int m = 0; m < 4; ++m) {
        const float* W = Ws[m];
        _Float16* f = frags + m * 4096;
        for (int idx = threadIdx.x; idx < 4096; idx += blockDim.x) {
            int h    = idx & 15;
            int lane = (idx >> 4) & 31;
            int c    = (idx >> 9) & 1;
            int ct   = (idx >> 10) & 3;
            int k = c * 32 + ((h >> 3) << 4) + ((lane >> 4) << 3) + (h & 7);
            int n = ct * 16 + (lane & 15);
            f[idx] = (_Float16)W[k * DIM + n];
        }
    }
}

// ------------------------------------------------------- edge scatter
// One thread per (edge, 4-float group): agg[dst] += x[src]
__global__ void scatter_add(const float* __restrict__ x,
                            const int* __restrict__ src,
                            const int* __restrict__ dst,
                            float* __restrict__ agg) {
    long long t = (long long)blockIdx.x * blockDim.x + threadIdx.x;
    if (t >= (long long)N_EDGES * 16) return;
    int e = (int)(t >> 4);
    int g = (int)(t & 15) * 4;
    int s = src[e], d = dst[e];
    float4 v = *(const float4*)(x + (size_t)s * DIM + g);
    float* ap = agg + (size_t)d * DIM + g;
    atomicAdd(ap + 0, v.x);
    atomicAdd(ap + 1, v.y);
    atomicAdd(ap + 2, v.z);
    atomicAdd(ap + 3, v.w);
}

// --------------------------------------------------- fused GIN MLP
// h = tanh( tanh((x+agg) @ W1 + b1) @ W2 + b2 )  via v_wmma_f32_16x16x32_f16
// Each wave owns a 16-row stripe, all 4 column tiles (64 cols).
// Weights (16 KB pre-swizzled fragment pair) staged to LDS via TDM/async DMA.
__global__ void __launch_bounds__(256)
gin_mlp(const float* __restrict__ xin, const float* agg,
        const _Float16* __restrict__ wpair,       // w1frag(4096h) ++ w2frag(4096h)
        const float* __restrict__ b1, const float* __restrict__ b2,
        float* hout, float* __restrict__ s1, float* __restrict__ s2) {
    __shared__ __align__(16) _Float16 wlds[2 * 4096];        // 16 KB
    __shared__ __align__(16) _Float16 hstage[8 * 16 * DIM];  // 16 KB

    size_t blk_row0 = (size_t)blockIdx.x * 8 * 16;
    if (blk_row0 < N_NODES)
        __builtin_prefetch(xin + blk_row0 * DIM + threadIdx.x * 16, 0, 1);

    int lane = threadIdx.x & 31;
    int wid  = threadIdx.x >> 5;

    // ---- stage the 16 KB weight-fragment pair into LDS -----------------
#if defined(STAGE_TDM5) || defined(STAGE_TDM6)
    if (wid == 0) {
        // Tensor DMA descriptor (CDNA5 ISA):
        // 1-D tile of 8192 2-byte elements: global wpair -> LDS wlds.
        unsigned long long ga = (unsigned long long)(uintptr_t)wpair;
        u32x4 g0;
        g0.x = 1u;                                         // count=1, user mode
        g0.y = (unsigned)(uintptr_t)wlds;                  // lds_addr (bytes)
        g0.z = (unsigned)(ga & 0xffffffffull);             // global_addr[31:0]
        g0.w = (unsigned)((ga >> 32) & 0x01ffffffull)      // global_addr[56:32]
             | 0x80000000u;                                // type=2 ("image")
        i32x8 g1;
        g1[0] = 0x00010000;          // data_size=1 (2 bytes); no multicast/pad
        g1[1] = (int)(8192u << 16);  // tensor_dim0[15:0]=8192
        g1[2] = 0x00010000;          // tensor_dim0[31:16]=0 | tensor_dim1=1
        g1[3] = (int)(8192u << 16);  // tile_dim0=8192
        g1[4] = 1;                   // tile_dim1=1, tile_dim2=0 (unused)
        g1[5] = 8192;                // tensor_dim0_stride=8192
        g1[6] = 0;
        g1[7] = 0;
        i32x4 z4 = {0, 0, 0, 0};     // groups 2/3: dims 2+ unused (2-D op)
#if defined(STAGE_TDM6)
        i32x8 z8 = {0, 0, 0, 0, 0, 0, 0, 0};
        __builtin_amdgcn_tensor_load_to_lds(g0, g1, z4, z4, z8, 0);
#else
        __builtin_amdgcn_tensor_load_to_lds(g0, g1, z4, z4, 0);
#endif
        __builtin_amdgcn_s_wait_tensorcnt(0);
    }
#elif defined(STAGE_ASYNC)
    {
        unsigned lbase = (unsigned)(uintptr_t)wlds;        // LDS byte offset
        for (int i = threadIdx.x; i < 1024; i += 256) {    // 1024 x 16 B
            as1_i32x4* gsrc = (as1_i32x4*)((uintptr_t)wpair + (size_t)i * 16);
            as3_i32x4* ldst = (as3_i32x4*)(lbase + i * 16);
            __builtin_amdgcn_global_load_async_to_lds_b128(gsrc, ldst, 0, 0);
        }
        __builtin_amdgcn_s_wait_asynccnt(0);
    }
#else
    {
        const float4* s4 = (const float4*)wpair;
        float4* d4 = (float4*)wlds;
        for (int i = threadIdx.x; i < 1024; i += 256) d4[i] = s4[i];
    }
#endif
    __syncthreads();

    int rt = blockIdx.x * 8 + wid;             // 16-row tile index
    if (rt * 16 >= N_NODES) return;            // wave-uniform (EXEC all-1s inside)

    int row0 = rt * 16;
    int r    = row0 + (lane & 15);
    int kb   = (lane >> 4) * 8;                // K-half base for A fragments

    // ---- load x+agg for this lane's A slice (4 runs of 8 contiguous floats)
    float xv[32];
    const float* xr = xin + (size_t)r * DIM;
    const float* ar = agg + (size_t)r * DIM;
#pragma unroll
    for (int g = 0; g < 4; ++g) {
        int kk = kb + g * 16;
        float4 a0 = *(const float4*)(xr + kk);
        float4 g0 = *(const float4*)(ar + kk);
        float4 a1 = *(const float4*)(xr + kk + 4);
        float4 g1 = *(const float4*)(ar + kk + 4);
        xv[g*8+0] = a0.x + g0.x; xv[g*8+1] = a0.y + g0.y;
        xv[g*8+2] = a0.z + g0.z; xv[g*8+3] = a0.w + g0.w;
        xv[g*8+4] = a1.x + g1.x; xv[g*8+5] = a1.y + g1.y;
        xv[g*8+6] = a1.z + g1.z; xv[g*8+7] = a1.w + g1.w;
    }
    v16h A0, A1;
#pragma unroll
    for (int h = 0; h < 16; ++h) {
        A0[h] = (_Float16)xv[h];
        A1[h] = (_Float16)xv[16 + h];
    }

    // ---- GEMM1: (x+agg) @ W1 + b1, tanh, stage to LDS for re-layout
    _Float16* hs = hstage + wid * 16 * DIM;
#pragma unroll
    for (int ct = 0; ct < 4; ++ct) {
        float bb = b1[ct * 16 + (lane & 15)];
        v8f c;
#pragma unroll
        for (int i = 0; i < 8; ++i) c[i] = bb;
        const _Float16* bp0 = wlds + ((ct * 2 + 0) * 32 + lane) * 16;
        const _Float16* bp1 = wlds + ((ct * 2 + 1) * 32 + lane) * 16;
        v16h B0, B1;
#pragma unroll
        for (int h = 0; h < 16; ++h) { B0[h] = bp0[h]; B1[h] = bp1[h]; }
        c = __builtin_amdgcn_wmma_f32_16x16x32_f16(false, A0, false, B0,
                                                   (short)0, c, false, false);
        c = __builtin_amdgcn_wmma_f32_16x16x32_f16(false, A1, false, B1,
                                                   (short)0, c, false, false);
#pragma unroll
        for (int i = 0; i < 8; ++i) {
            int rr = i + 8 * (lane >> 4);
            hs[rr * DIM + ct * 16 + (lane & 15)] = (_Float16)fast_tanh(c[i]);
        }
    }

    // ---- re-gather A fragments for GEMM2 from LDS (transposed distribution)
    int r2 = lane & 15;
    v16h A20, A21;
#pragma unroll
    for (int h = 0; h < 16; ++h) {
        int k0 = kb + ((h >> 3) << 4) + (h & 7);
        A20[h] = hs[r2 * DIM + k0];
        A21[h] = hs[r2 * DIM + 32 + k0];
    }

    // ---- GEMM2 + tanh, write pre-BN activations, accumulate BN stats
#pragma unroll
    for (int ct = 0; ct < 4; ++ct) {
        int col = ct * 16 + (lane & 15);
        float bb = b2[col];
        v8f c;
#pragma unroll
        for (int i = 0; i < 8; ++i) c[i] = bb;
        const _Float16* bp0 = wlds + 4096 + ((ct * 2 + 0) * 32 + lane) * 16;
        const _Float16* bp1 = wlds + 4096 + ((ct * 2 + 1) * 32 + lane) * 16;
        v16h B0, B1;
#pragma unroll
        for (int h = 0; h < 16; ++h) { B0[h] = bp0[h]; B1[h] = bp1[h]; }
        c = __builtin_amdgcn_wmma_f32_16x16x32_f16(false, A20, false, B0,
                                                   (short)0, c, false, false);
        c = __builtin_amdgcn_wmma_f32_16x16x32_f16(false, A21, false, B1,
                                                   (short)0, c, false, false);
        float ls = 0.0f, lq = 0.0f;
#pragma unroll
        for (int i = 0; i < 8; ++i) {
            float v = fast_tanh(c[i]);
            int rr = row0 + i + 8 * (lane >> 4);
            hout[(size_t)rr * DIM + col] = v;
            ls += v;
            lq += v * v;
        }
        atomicAdd(&s1[col], ls);
        atomicAdd(&s2[col], lq);
    }
}

// --------------------------------------------------- BN finalize (64 cols)
__global__ void bn_finalize(const float* __restrict__ s1, const float* __restrict__ s2,
                            const float* __restrict__ gamma, const float* __restrict__ beta,
                            float* __restrict__ scale, float* __restrict__ shift) {
    int c = threadIdx.x;
    if (c >= DIM) return;
    float inv_n = 1.0f / (float)N_NODES;
    float mu  = s1[c] * inv_n;
    float var = s2[c] * inv_n - mu * mu;
    float sc  = gamma[c] * rsqrtf(var + BN_EPS);
    scale[c] = sc;
    shift[c] = beta[c] - mu * sc;
}

// ------------------------------------- BN apply + per-graph add-pool
__global__ void bn_pool(const float* __restrict__ h,
                        const float* __restrict__ scale, const float* __restrict__ shift,
                        const int* __restrict__ batch, float* __restrict__ out,
                        int colOff, float* __restrict__ xnext) {
    long long t = (long long)blockIdx.x * blockDim.x + threadIdx.x;
    if (t >= (long long)N_NODES * 32) return;
    int node = (int)(t >> 5);
    int cp   = (int)(t & 31) * 2;
    float2 hv = *(const float2*)(h + (size_t)node * DIM + cp);
    float y0 = hv.x * scale[cp]     + shift[cp];
    float y1 = hv.y * scale[cp + 1] + shift[cp + 1];
    if (xnext) {
        xnext[(size_t)node * DIM + cp]     = y0;
        xnext[(size_t)node * DIM + cp + 1] = y1;
    }
    int g = batch[node];
    atomicAdd(&out[(size_t)g * 128 + colOff + cp],     y0);
    atomicAdd(&out[(size_t)g * 128 + colOff + cp + 1], y1);
}

// ================================================================ launch
extern "C" void kernel_launch(void* const* d_in, const int* in_sizes, int n_in,
                              void* d_out, int out_size, void* d_ws, size_t ws_size,
                              hipStream_t stream) {
    const float* x       = (const float*)d_in[0];
    const float* W1_0    = (const float*)d_in[1];
    const float* b1_0    = (const float*)d_in[2];
    const float* W2_0    = (const float*)d_in[3];
    const float* b2_0    = (const float*)d_in[4];
    const float* gamma_0 = (const float*)d_in[5];
    const float* beta_0  = (const float*)d_in[6];
    const float* W1_1    = (const float*)d_in[7];
    const float* b1_1    = (const float*)d_in[8];
    const float* W2_1    = (const float*)d_in[9];
    const float* b2_1    = (const float*)d_in[10];
    const float* gamma_1 = (const float*)d_in[11];
    const float* beta_1  = (const float*)d_in[12];
    const int*   eidx    = (const int*)d_in[13];
    const int*   batch   = (const int*)d_in[14];
    const int*   src = eidx;
    const int*   dst = eidx + N_EDGES;
    float* out = (float*)d_out;

    char* ws = (char*)d_ws;
    float* buf0 = (float*)ws;                            // 25.6 MB: agg / pre-BN h
    float* buf1 = (float*)(ws + 25600000);               // 25.6 MB: layer-0 output x0
    float* stats = (float*)(ws + 51200000);              // 512 floats
    _Float16* frags = (_Float16*)(ws + 51200000 + 4096); // 4 x 4096 halves

    float* s1_0 = stats,         *s2_0 = stats + 64;
    float* s1_1 = stats + 128,   *s2_1 = stats + 192;
    float* scale0 = stats + 256, *shift0 = stats + 320;
    float* scale1 = stats + 384, *shift1 = stats + 448;

    const int nodeF = N_NODES * DIM;                     // 6,400,000
    dim3 blk(256);

    zero_f32<<<(NUM_GRAPHS * 128 + 255) / 256, blk, 0, stream>>>(out, NUM_GRAPHS * 128);
    zero_f32<<<1, blk, 0, stream>>>(stats, 256);
    prep_weights<<<1, blk, 0, stream>>>(W1_0, W2_0, W1_1, W2_1, frags);

    // ---------------- layer 0
    zero_f32<<<(nodeF + 255) / 256, blk, 0, stream>>>(buf0, nodeF);
    scatter_add<<<(N_EDGES * 16) / 256, blk, 0, stream>>>(x, src, dst, buf0);
    gin_mlp<<<(N_NODES / 16 + 7) / 8, blk, 0, stream>>>(
        x, buf0, frags, b1_0, b2_0, buf0, s1_0, s2_0);
    bn_finalize<<<1, 64, 0, stream>>>(s1_0, s2_0, gamma_0, beta_0, scale0, shift0);
    bn_pool<<<(N_NODES * 32) / 256, blk, 0, stream>>>(
        buf0, scale0, shift0, batch, out, 0, buf1);

    // ---------------- layer 1
    zero_f32<<<(nodeF + 255) / 256, blk, 0, stream>>>(buf0, nodeF);
    scatter_add<<<(N_EDGES * 16) / 256, blk, 0, stream>>>(buf1, src, dst, buf0);
    gin_mlp<<<(N_NODES / 16 + 7) / 8, blk, 0, stream>>>(
        buf1, buf0, frags + 8192, b1_1, b2_1, buf0, s1_1, s2_1);
    bn_finalize<<<1, 64, 0, stream>>>(s1_1, s2_1, gamma_1, beta_1, scale1, shift1);
    bn_pool<<<(N_NODES * 32) / 256, blk, 0, stream>>>(
        buf0, scale1, shift1, batch, out, 64, nullptr);
}
